// GCN_53403623358580
// MI455X (gfx1250) — matrix operand order repaired and checked by reading, not compiled, not run on hardware
//
#include <hip/hip_runtime.h>
#include <hip/hip_bf16.h>

#define N_NODES 100000
#define N_EDGES 3200000
#define D_IN    768
#define D_H1    256
#define D_H2    128
#define HID     32
#define NC      3

typedef __bf16 bf16_t;
typedef bf16_t v16bf __attribute__((ext_vector_type(16)));
typedef bf16_t v8bf  __attribute__((ext_vector_type(8)));
typedef float  v8f   __attribute__((ext_vector_type(8)));

#if __has_builtin(__builtin_amdgcn_sched_group_barrier)
#define SGB(mask, size, id) __builtin_amdgcn_sched_group_barrier(mask, size, id)
#else
#define SGB(mask, size, id)
#endif
// sched_group_barrier masks: 0x100 = DS read, 0x008 = MFMA/WMMA

// ---------------------------------------------------------------------------
// Prep 1: W_emb0 fp32 [768][256] -> bf16 transposed [256][768] (L2-resident)
// ---------------------------------------------------------------------------
__global__ void k_w0_to_bf16t(const float* __restrict__ w0, bf16_t* __restrict__ wt) {
  int idx = blockIdx.x * 256 + threadIdx.x;
  if (idx < D_IN * D_H1) {
    int k = idx / D_H1, c = idx % D_H1;
    wt[(size_t)c * D_IN + k] = (bf16_t)w0[idx];
  }
}

// ---------------------------------------------------------------------------
// Prep 2: collapse the linear tail.
//   Wlc = W_lin @ W_conv                      [128,3]
//   W2c = W_emb1 @ Wlc                        [256,3]
//   b2c = b_emb1 @ Wlc + b_lin @ W_conv       [3]
// ---------------------------------------------------------------------------
__global__ void k_fuse_tail(const float* __restrict__ W1, const float* __restrict__ b1,
                            const float* __restrict__ Wl, const float* __restrict__ bl,
                            const float* __restrict__ Wc,
                            float* __restrict__ W2c, float* __restrict__ b2c) {
  __shared__ float sWc[HID * NC];    // 96
  __shared__ float sWlc[D_H2 * NC];  // 384
  const int t = threadIdx.x;
  if (t < HID * NC) sWc[t] = Wc[t];
  __syncthreads();
  for (int i = t; i < D_H2 * NC; i += 256) {
    int r = i / NC, c = i % NC;
    float s = 0.f;
    for (int j = 0; j < HID; ++j) s += Wl[r * HID + j] * sWc[j * NC + c];
    sWlc[i] = s;
  }
  __syncthreads();
  for (int i = t; i < D_H1 * NC; i += 256) {
    int r = i / NC, c = i % NC;
    float s = 0.f;
    for (int k = 0; k < D_H2; ++k) s += W1[r * D_H2 + k] * sWlc[k * NC + c];
    W2c[i] = s;
  }
  if (t < NC) {
    float s = 0.f;
    for (int k = 0; k < D_H2; ++k) s += b1[k] * sWlc[k * NC + t];
    for (int j = 0; j < HID; ++j) s += bl[j] * sWc[j * NC + t];
    b2c[t] = s;
  }
}

// ---------------------------------------------------------------------------
// Main fused kernel: h4 = relu(X @ W_emb0 + b0) @ W2c + b2c   -> [N,3]
// 64 rows x 256 cols per workgroup; 8 waves (4x2); each wave: 8 WMMA tiles.
//  - X chunk k+1 staged into VGPRs while chunk k computes (reg double-buffer)
//  - W chunk copied with global_load_async_to_lds_b128 (ASYNCcnt path)
//  - sched_group_barrier forces all 18 DS reads ahead of the 8 WMMAs so
//    waitcnts stagger instead of fully draining per matrix op
// ---------------------------------------------------------------------------
#define BM  64
#define KC  64
#define LDA 72    // bf16 stride for sA (16B-aligned rows)
#define LDB 72    // bf16 stride for sB (col-major [col][k])
#define LDH 260   // f32 stride for sH

__launch_bounds__(256)
__global__ void k_gcn_mlp(const float* __restrict__ x,
                          const bf16_t* __restrict__ wt,   // [256][768] bf16
                          const float* __restrict__ b0,
                          const float* __restrict__ w2c,
                          const float* __restrict__ b2c,
                          float* __restrict__ h4) {
  __shared__ bf16_t sA[BM][LDA];        // 9.2 KB  : X tile (64 rows x 64 K)
  __shared__ bf16_t sB[D_H1][LDB];      // 36.9 KB : W tile col-major (256 x 64)
  __shared__ float  sH[BM][LDH];        // 66.6 KB : h1 block
  __shared__ float  sW2c[D_H1 * NC + NC];

  const int t    = threadIdx.x;
  const int row0 = blockIdx.x * BM;

  for (int i = t; i < D_H1 * NC; i += 256) sW2c[i] = w2c[i];
  if (t < NC) sW2c[D_H1 * NC + t] = b2c[t];

  const int lane = t & 31;
  const int wid  = t >> 5;
  const int wm   = wid & 3;    // 16-row group
  const int wn   = wid >> 2;   // 128-col group (0..1)
  const int l15  = lane & 15;
  const int hi   = lane >> 4;  // half-wave select

  v8f acc[8] = {};

  // register staging for the X tile (fp32, converted to bf16 at LDS store)
  float4 rx[4];
  auto load_x = [&](int kc) {
    #pragma unroll
    for (int i = 0; i < 4; ++i) {
      int li = t + i * 256;          // 0..1023
      int r  = li >> 4;
      int kq = li & 15;
      int grow = row0 + r;
      rx[i] = make_float4(0.f, 0.f, 0.f, 0.f);
      if (grow < N_NODES)
        rx[i] = *(const float4*)(x + (size_t)grow * D_IN + kc + kq * 4);
    }
  };
  load_x(0);

  for (int kc = 0; kc < D_IN; kc += KC) {
    __syncthreads();
    // --- commit staged X regs to LDS (fp32 -> bf16) ---
    #pragma unroll
    for (int i = 0; i < 4; ++i) {
      int li = t + i * 256;
      int r  = li >> 4;
      int kq = li & 15;
      bf16_t* dst = &sA[r][kq * 4];
      dst[0] = (bf16_t)rx[i].x; dst[1] = (bf16_t)rx[i].y;
      dst[2] = (bf16_t)rx[i].z; dst[3] = (bf16_t)rx[i].w;
    }
    // --- async copy W chunk (bf16, pre-transposed) straight into LDS ---
    #pragma unroll
    for (int i = 0; i < 8; ++i) {
      int li  = t + i * 256;         // 0..2047, 16B per unit
      int c   = li >> 3;
      int seg = li & 7;
      unsigned lds_off = (unsigned)(size_t)&sB[c][seg * 8];
      unsigned long long ga =
          (unsigned long long)(const void*)(wt + (size_t)c * D_IN + kc + seg * 8);
      asm volatile("global_load_async_to_lds_b128 %0, %1, off"
                   :: "v"(lds_off), "v"(ga) : "memory");
    }
    // --- kick off next X chunk's global loads; they overlap the WMMAs ---
    if (kc + KC < D_IN) load_x(kc + KC);

    asm volatile("s_wait_asynccnt 0x0" ::: "memory");
    __syncthreads();

    // --- 2 K-steps of 32 per 64-K chunk ---
    #pragma unroll
    for (int s = 0; s < 2; ++s) {
      // A fragment (contiguous K, matching B): lanes 0-15 K 0..15;
      // lanes 16-31 K 16..31 of the same rows.
      const int ar = wm * 16 + l15;
      const int kb = s * 32 + hi * 16;
      v8bf alo = *(const v8bf*)&sA[ar][kb];
      v8bf ahi = *(const v8bf*)&sA[ar][kb + 8];
      v16bf a;
      #pragma unroll
      for (int i = 0; i < 8; ++i) { a[i] = alo[i]; a[i + 8] = ahi[i]; }

      const int kbB = s * 32 + hi * 16;
      // preload ALL 8 B fragments into distinct register tuples
      v16bf bfrag[8];
      #pragma unroll
      for (int j = 0; j < 8; ++j) {
        const int col = wn * 128 + j * 16 + l15;
        bfrag[j] = *(const v16bf*)&sB[col][kbB];
      }
      #pragma unroll
      for (int j = 0; j < 8; ++j)
        acc[j] = __builtin_amdgcn_wmma_f32_16x16x32_bf16(
            false, a, false, bfrag[j], (short)0, acc[j], false, false);

      // force schedule: 18 DS reads (A:2 + B:16) first, then 8 WMMAs.
      // waitcnts then stagger (dscnt 14,12,...,0) instead of draining per op.
      SGB(0x100, 18, 0);
      SGB(0x008, 8, 0);
    }
  }

  // --- epilogue 1: bias + ReLU into LDS (C layout: M = r + 8*hi, N = l15) ---
  __syncthreads();
  #pragma unroll
  for (int tn = 0; tn < 8; ++tn) {
    const int n    = wn * 128 + tn * 16 + l15;
    const float bv = b0[n];
    #pragma unroll
    for (int r = 0; r < 8; ++r) {
      const int m = wm * 16 + hi * 8 + r;
      float v = acc[tn][r] + bv;
      sH[m][n] = v > 0.f ? v : 0.f;
    }
  }
  __syncthreads();

  // --- epilogue 2: [64,256] @ [256,3] + b2c, straight out of LDS ---
  if (t < BM * NC) {
    const int r = t / NC, c = t % NC;
    float s = sW2c[D_H1 * NC + c];
    #pragma unroll 4
    for (int k = 0; k < D_H1; ++k) s += sH[r][k] * sW2c[k * NC + c];
    int grow = row0 + r;
    if (grow < N_NODES) h4[(size_t)grow * NC + c] = s;
  }
}

// ---------------------------------------------------------------------------
// Graph aggregation (all L2-resident: h4 1.2MB, deg/dinv 0.8MB)
// ---------------------------------------------------------------------------
__global__ void k_deg_init(float* __restrict__ deg) {
  int i = blockIdx.x * 256 + threadIdx.x;
  if (i < N_NODES) deg[i] = 1.0f;  // self loop
}

__global__ void k_deg_edges(const int* __restrict__ ei, float* __restrict__ deg) {
  int e = blockIdx.x * 256 + threadIdx.x;
  if (e < N_EDGES) atomicAdd(&deg[ei[N_EDGES + e]], 1.0f);
}

__global__ void k_self(const float* __restrict__ deg, float* __restrict__ dinv,
                       const float* __restrict__ h4, const float* __restrict__ bc,
                       float* __restrict__ out) {
  int i = blockIdx.x * 256 + threadIdx.x;
  if (i < N_NODES) {
    float d  = deg[i];
    float di = d > 0.f ? rsqrtf(d) : 0.f;
    dinv[i] = di;
    float w = di * di;  // self-loop norm
    #pragma unroll
    for (int c = 0; c < NC; ++c)
      out[(size_t)i * NC + c] = h4[(size_t)i * NC + c] * w + bc[c];
  }
}

__global__ void k_scatter(const int* __restrict__ ei, const float* __restrict__ dinv,
                          const float* __restrict__ h4, float* __restrict__ out) {
  int e = blockIdx.x * 256 + threadIdx.x;
  if (e < N_EDGES) {
    int s = ei[e], d = ei[N_EDGES + e];
    float w = dinv[s] * dinv[d];
    #pragma unroll
    for (int c = 0; c < NC; ++c)
      atomicAdd(&out[(size_t)d * NC + c], h4[(size_t)s * NC + c] * w);
  }
}

// ---------------------------------------------------------------------------
extern "C" void kernel_launch(void* const* d_in, const int* in_sizes, int n_in,
                              void* d_out, int out_size, void* d_ws, size_t ws_size,
                              hipStream_t stream) {
  const float* x  = (const float*)d_in[0];
  const int*   ei = (const int*)d_in[1];   // edge_index [2, E]
  const float* W0 = (const float*)d_in[2];
  const float* b0 = (const float*)d_in[3];
  const float* W1 = (const float*)d_in[4];
  const float* b1 = (const float*)d_in[5];
  const float* Wl = (const float*)d_in[6];
  const float* bl = (const float*)d_in[7];
  const float* Wc = (const float*)d_in[8];
  const float* bc = (const float*)d_in[9];
  float* out = (float*)d_out;

  char* ws = (char*)d_ws;
  bf16_t* wt  = (bf16_t*)(ws);             // 393,216 B
  float* W2c  = (float*)(ws + 393216);     //   3,072 B
  float* b2c  = (float*)(ws + 396288);     //     256 B
  float* h4   = (float*)(ws + 396544);     // 1,200,000 B
  float* deg  = (float*)(ws + 1596544);    //   400,000 B
  float* dinv = (float*)(ws + 1996544);    //   400,000 B

  k_w0_to_bf16t<<<(D_IN * D_H1 + 255) / 256, 256, 0, stream>>>(W0, wt);
  k_fuse_tail<<<1, 256, 0, stream>>>(W1, b1, Wl, bl, Wc, W2c, b2c);
  k_gcn_mlp<<<(N_NODES + BM - 1) / BM, 256, 0, stream>>>(x, wt, b0, W2c, b2c, h4);
  k_deg_init<<<(N_NODES + 255) / 256, 256, 0, stream>>>(deg);
  k_deg_edges<<<(N_EDGES + 255) / 256, 256, 0, stream>>>(ei, deg);
  k_self<<<(N_NODES + 255) / 256, 256, 0, stream>>>(deg, dinv, h4, bc, out);
  k_scatter<<<(N_EDGES + 255) / 256, 256, 0, stream>>>(ei, dinv, h4, out);
}
